// VideoModelCoord_18459769438237
// MI455X (gfx1250) — compile-verified, matrix operand order checked by reading, not run on hardware
//
#include <hip/hip_runtime.h>
#include <hip/hip_bf16.h>

// ---------------------------------------------------------------------------
// CDNA5 (gfx1250, wave32) VideoModel forward pass.
// - All dense layers via one WMMA GEMM kernel (v_wmma_f32_16x16x32_f16):
//     block tile 64(M) x 256(N) x 32(K), 8 waves, wave tile 32x64 (2x4 frags
//     -> 8 WMMA / K-tile / wave).
// - Weights pre-converted once per launch to zero-padded f16 in d_ws; B tiles
//   staged LDS via global_load_async_to_lds_b128 (ASYNCcnt + s_wait_asynccnt).
// - A tiles: vectorized f32 loads + packed f16 cvt + single ds_store_b128.
// - Dilated (1,3) convs = 3 shifted accumulate-GEMMs with repacked tap wts.
//
// Assumed d_in flattening (setup_inputs() dict insertion order, depth-first):
//   0 box_input; 1..6 coord; 7..12 mlp1; 13..18 mlp2; 19..24 mlp3; 25..30 mlp4
//   31 edge2node; 32 Bk_verb; 33 Bk_prep; 34/35 conv_vd w,b; 36/37 conv_pd w,b
//   38..55 V_stage; 56..73 P_stage; 74..91 nohand_stage (w0,b0,4x{wd,bd,w1,b1})
//   92..97 V_fusion; 98..103 P_fusion; 104/105 nohand_fusion w1,w2
//   106..111 mlp5; 112..117 mlp6; 118/119 mlp7 w1,w2
//   120..125 cls; 126..131 cls_v; 132..137 cls_p (w1,b1,w2,b2,w3,b3)
// ---------------------------------------------------------------------------

#define BATCH   512
#define NBOX    4
#define NFRM    16
#define NEDGE   12
#define DIM     256
#define MX      (BATCH*NBOX*NFRM)    /* 32768 node rows */
#define ME      (BATCH*NEDGE*NFRM)   /* 98304 edge rows */

__device__ __constant__ int c_SEND[NEDGE] = {0,0,0,1,1,1,2,2,2,3,3,3};
__device__ __constant__ int c_RECV[NEDGE] = {1,2,3,0,2,3,0,1,3,0,1,2};

typedef _Float16 half8   __attribute__((ext_vector_type(8)));
typedef _Float16 half16  __attribute__((ext_vector_type(16)));
typedef float    float8  __attribute__((ext_vector_type(8)));
typedef float    float4v __attribute__((ext_vector_type(4)));

#define GBM 64
#define GBN 256
#define GBK 32
#define LDSW 40   /* padded LDS row stride in halfs (80B, 16B-aligned rows) */

// ---------------------------------------------------------------------------
// C[M,N] = act( A[M,K](f32) * Wf16[Npad,Kpad]^T + bias [+ C] )
// Optional temporal row shift of A (dilated conv taps):
//   Tdim>0: t=m%Tdim; valid iff 0<=t+tshift<Tdim; row=m+tshift
// ---------------------------------------------------------------------------
__global__ __launch_bounds__(256)
void gemm_wmma_kernel(const float* __restrict__ A, const _Float16* __restrict__ Bw,
                      const float* __restrict__ bias, float* __restrict__ C,
                      int M, int N, int K, int lda, int ldb,
                      int Tdim, int tshift, int act, int beta)
{
    __shared__ __align__(16) _Float16 As[2][GBM][LDSW];
    __shared__ __align__(16) _Float16 Bs[2][GBN][LDSW];

    const int tid   = threadIdx.x;
    const int mBase = blockIdx.y * GBM;
    const int nBase = blockIdx.x * GBN;
    const int lane  = tid & 31;
    const int w     = tid >> 5;
    const int wm    = (w & 1) * 32;     // wave M offset (2 rows of waves)
    const int wn    = (w >> 1) * 64;    // wave N offset (4 cols of waves)
    const int l16   = lane & 15;
    const int khalf = lane >> 4;

    float8 acc[2][4];
#pragma unroll
    for (int i = 0; i < 2; ++i)
#pragma unroll
        for (int j = 0; j < 4; ++j) {
            float8 z = {0.f,0.f,0.f,0.f,0.f,0.f,0.f,0.f};
            acc[i][j] = z;
        }

    const int nk = (K + GBK - 1) / GBK;

    auto loadTile = [&](int kt, int bufi) {
        const int k0 = kt * GBK;
        // ---- B tile: 256 rows x 32 halfs, async DMA to LDS (16B x 4/thr) --
        {
            const _Float16* bsrc = Bw + (size_t)nBase * ldb + k0;
#pragma unroll
            for (int i = 0; i < 4; ++i) {
                const int c = tid + 256 * i;
                const int r = c >> 2, q = c & 3;
                unsigned lds = (unsigned)(uintptr_t)(&Bs[bufi][r][q * 8]);
                unsigned long long g =
                    (unsigned long long)(uintptr_t)(bsrc + (size_t)r * ldb + q * 8);
                asm volatile("global_load_async_to_lds_b128 %0, %1, off"
                             :: "v"(lds), "v"(g) : "memory");
            }
        }
        // ---- A tile: 64 rows x 32 f32, 8 elems/thread, one b128 LDS store -
        {
            const int r  = tid >> 2;
            const int kb = (tid & 3) * 8;
            const int m  = mBase + r;
            bool mok = (m < M);
            int  row = m;
            if (Tdim > 0) {
                const int t  = m % Tdim;
                const int tt = t + tshift;
                mok = mok && (tt >= 0) && (tt < Tdim);
                row = m + tshift;
            }
            const float* ap = A + (size_t)row * lda + k0 + kb;
            half8 h;
            if (mok && (k0 + kb + 8) <= K) {
                if ((k0 + GBK) < K) __builtin_prefetch(ap + GBK, 0, 1);
                const float4v* ap4 = reinterpret_cast<const float4v*>(ap);
                float4v a0 = ap4[0], a1 = ap4[1];
#pragma unroll
                for (int j = 0; j < 4; ++j) {
                    h[j]     = (_Float16)a0[j];
                    h[4 + j] = (_Float16)a1[j];
                }
            } else {
#pragma unroll
                for (int j = 0; j < 8; ++j) {
                    float v = (mok && (k0 + kb + j) < K) ? ap[j] : 0.0f;
                    h[j] = (_Float16)v;
                }
            }
            *reinterpret_cast<half8*>(&As[bufi][r][kb]) = h;
        }
    };

    loadTile(0, 0);
    int buf = 0;
    for (int kt = 0; kt < nk; ++kt) {
        asm volatile("s_wait_asynccnt 0x0" ::: "memory");  // B tile in LDS
        __syncthreads();
        if (kt + 1 < nk) loadTile(kt + 1, buf ^ 1);

        half16 af[2], bf[4];
#pragma unroll
        for (int im = 0; im < 2; ++im) {
            const int m = wm + im * 16 + l16;
            const half8* p0 = reinterpret_cast<const half8*>(&As[buf][m][khalf * 8]);
            const half8* p1 = reinterpret_cast<const half8*>(&As[buf][m][16 + khalf * 8]);
            af[im] = __builtin_shufflevector(p0[0], p1[0],
                        0,1,2,3,4,5,6,7,8,9,10,11,12,13,14,15);
        }
#pragma unroll
        for (int in = 0; in < 4; ++in) {
            const int n = wn + in * 16 + l16;
            const half8* p0 = reinterpret_cast<const half8*>(&Bs[buf][n][khalf * 8]);
            const half8* p1 = reinterpret_cast<const half8*>(&Bs[buf][n][16 + khalf * 8]);
            bf[in] = __builtin_shufflevector(p0[0], p1[0],
                        0,1,2,3,4,5,6,7,8,9,10,11,12,13,14,15);
        }
#pragma unroll
        for (int im = 0; im < 2; ++im)
#pragma unroll
            for (int in = 0; in < 4; ++in)
                acc[im][in] = __builtin_amdgcn_wmma_f32_16x16x32_f16(
                    false, af[im], false, bf[in], (short)0, acc[im][in],
                    false, false);
        buf ^= 1;
    }

    // ---- epilogue: lane(l16)=n, vgpr r + 8*khalf = m ---------------------
#pragma unroll
    for (int im = 0; im < 2; ++im)
#pragma unroll
        for (int in = 0; in < 4; ++in) {
            const int n = nBase + wn + in * 16 + l16;
#pragma unroll
            for (int r = 0; r < 8; ++r) {
                const int m = mBase + wm + im * 16 + r + 8 * khalf;
                if (m < M && n < N) {
                    const long idx = (long)m * N + n;
                    float v = acc[im][in][r];
                    if (bias) v += bias[n];
                    if (beta) v += C[idx];
                    if (act == 1)      v = fmaxf(v, 0.0f);
                    else if (act == 2) v = (v > 0.0f) ? v : (__expf(v) - 1.0f);
                    C[idx] = v;
                }
            }
        }
}

// ---------------------------------------------------------------------------
// Weight conversion: f32 (strided) -> zero-padded f16 [Npad, Kpad]
// ---------------------------------------------------------------------------
__global__ __launch_bounds__(256)
void cvt_w_f16_kernel(const float* __restrict__ in, _Float16* __restrict__ out,
                      int N, int K, int Kpad, int ldRow, int ldCol, int total)
{
    const int idx = blockIdx.x * 256 + threadIdx.x;
    if (idx >= total) return;
    const int n = idx / Kpad, k = idx - n * Kpad;
    out[idx] = (n < N && k < K) ? (_Float16)in[(size_t)n * ldRow + (size_t)k * ldCol]
                                : (_Float16)0.0f;
}

// ---------------------------------------------------------------------------
// BatchNorm (training mode)
// ---------------------------------------------------------------------------
__global__ __launch_bounds__(256)
void bn_stats_kernel(const float* __restrict__ x, int M, int C,
                     float* __restrict__ mean, float* __restrict__ rstd)
{
    __shared__ float s1[256], s2[256];
    const int c = blockIdx.x, tid = threadIdx.x;
    float s = 0.f, q = 0.f;
    for (int i = tid; i < M; i += 256) {
        float v = x[(size_t)i * C + c];
        s += v; q += v * v;
    }
    s1[tid] = s; s2[tid] = q;
    __syncthreads();
    for (int off = 128; off; off >>= 1) {
        if (tid < off) { s1[tid] += s1[tid + off]; s2[tid] += s2[tid + off]; }
        __syncthreads();
    }
    if (tid == 0) {
        float m = s1[0] / (float)M;
        float v = s2[0] / (float)M - m * m;
        mean[c] = m;
        rstd[c] = rsqrtf(v + 1e-5f);
    }
}

__global__ __launch_bounds__(256)
void bn_apply_kernel(float* __restrict__ x, const float* __restrict__ mean,
                     const float* __restrict__ rstd, const float* __restrict__ g,
                     const float* __restrict__ b, int total, int C, int act)
{
    const int idx = blockIdx.x * 256 + threadIdx.x;
    if (idx >= total) return;
    const int c = idx % C;
    float v = (x[idx] - mean[c]) * rstd[c] * g[c] + b[c];
    if (act == 1) v = fmaxf(v, 0.0f);
    x[idx] = v;
}

// ---------------------------------------------------------------------------
// Graph / elementwise helpers
// ---------------------------------------------------------------------------
__global__ __launch_bounds__(256)
void transpose_box_kernel(const float* __restrict__ in, float* __restrict__ out)
{
    const int idx = blockIdx.x * 256 + threadIdx.x;
    if (idx >= BATCH * NBOX * NFRM * 4) return;
    const int c = idx & 3, t = (idx >> 2) & 15, n = (idx >> 6) & 3, b = idx >> 8;
    out[idx] = in[(((b * NFRM + t) * NBOX + n) << 2) + c];
}

__global__ __launch_bounds__(256)
void node2edge_kernel(const float* __restrict__ x, float* __restrict__ e, int total)
{
    const int idx = blockIdx.x * 256 + threadIdx.x;
    if (idx >= total) return;                              // total = ME*512
    const int col = idx & 511, row = idx >> 9;
    const int t = row & 15, eix = (row >> 4) % NEDGE, b = row / (NFRM * NEDGE);
    const int node = (col < DIM) ? c_SEND[eix] : c_RECV[eix];
    e[idx] = x[(((b * NBOX + node) * NFRM + t) << 8) + (col & 255)];
}

__global__ __launch_bounds__(256)
void edge2node_kernel(const float* __restrict__ e2n, const float* __restrict__ E,
                      float* __restrict__ out, int total)
{
    const int idx = blockIdx.x * 256 + threadIdx.x;
    if (idx >= total) return;                              // total = MX*256
    const int c = idx & 255, row = idx >> 8;
    const int t = row & 15, v = (row >> 4) & 3, b = row >> 6;
    float s = 0.f;
#pragma unroll
    for (int ei = 0; ei < NEDGE; ++ei)
        s += e2n[v * NEDGE + ei] * E[(((b * NEDGE + ei) * NFRM + t) << 8) + c];
    out[idx] = s * (1.0f / (NBOX - 1));
}

__global__ __launch_bounds__(256)
void bkmix_kernel(const float* __restrict__ Bk, const float* __restrict__ in,
                  float* __restrict__ out, int total)
{
    const int idx = blockIdx.x * 256 + threadIdx.x;
    if (idx >= total) return;                              // total = ME*256
    const int c = idx & 255, row = idx >> 8;
    const int t = row & 15, ei = (row >> 4) % NEDGE, b = row / (NFRM * NEDGE);
    float s = 0.f;
#pragma unroll
    for (int f = 0; f < NEDGE; ++f)
        s += Bk[ei * NEDGE + f] * in[(((b * NEDGE + f) * NFRM + t) << 8) + c];
    out[idx] = s;
}

__global__ __launch_bounds__(256)
void concat2_kernel(const float* __restrict__ a, const float* __restrict__ b,
                    float* __restrict__ o, int M, int C1, int C2)
{
    const int C = C1 + C2;
    const int idx = blockIdx.x * 256 + threadIdx.x;
    if (idx >= M * C) return;
    const int r = idx / C, c = idx - r * C;
    o[idx] = (c < C1) ? a[(size_t)r * C1 + c] : b[(size_t)r * C2 + (c - C1)];
}

__global__ __launch_bounds__(256)
void mean_group_kernel(const float* __restrict__ in, float* __restrict__ o,
                       int Bq, int G, int C)
{
    const int idx = blockIdx.x * 256 + threadIdx.x;
    if (idx >= Bq * C) return;
    const int b = idx / C, c = idx - b * C;
    float s = 0.f;
    for (int g = 0; g < G; ++g) s += in[(size_t)(b * G + g) * C + c];
    o[idx] = s / (float)G;
}

__global__ __launch_bounds__(256)
void bias_relu_kernel(float* __restrict__ y, const float* __restrict__ b,
                      int total, int C)
{
    const int idx = blockIdx.x * 256 + threadIdx.x;
    if (idx >= total) return;
    y[idx] = fmaxf(y[idx] + b[idx % C], 0.0f);
}

__global__ __launch_bounds__(256)
void add_inplace_kernel(float* __restrict__ y, const float* __restrict__ x, int total)
{
    const int idx = blockIdx.x * 256 + threadIdx.x;
    if (idx < total) y[idx] += x[idx];
}

// ---------------------------------------------------------------------------
static inline int ceilDiv(int a, int b) { return (a + b - 1) / b; }

struct WB { const _Float16* p; int ldb; };

extern "C" void kernel_launch(void* const* d_in, const int* in_sizes, int n_in,
                              void* d_out, int out_size, void* d_ws, size_t ws_size,
                              hipStream_t stream)
{
    (void)in_sizes; (void)n_in; (void)out_size; (void)ws_size;
    const float* P[138];
    for (int i = 0; i < 138; ++i) P[i] = (const float*)d_in[i];
    const float* box = P[0];

    // ---- bump allocator over d_ws ---------------------------------------
    size_t off = 0;
    auto allocBytes = [&](size_t bytes) -> void* {
        void* p = (char*)d_ws + off;
        off += (bytes + 255) & ~(size_t)255;
        return p;
    };
    auto alloc = [&](size_t elems) -> float* {
        return (float*)allocBytes(elems * sizeof(float));
    };

    // ---- weight conversion pass: f32 -> padded f16 -----------------------
    auto cvtW = [&](const float* wsrc, int N, int K, int ldRow, int ldCol) -> WB {
        const int Npad = ceilDiv(N, GBN) * GBN;
        const int Kpad = ceilDiv(K, GBK) * GBK;
        _Float16* o = (_Float16*)allocBytes((size_t)Npad * Kpad * sizeof(_Float16));
        const int total = Npad * Kpad;
        cvt_w_f16_kernel<<<ceilDiv(total, 256), 256, 0, stream>>>(
            wsrc, o, N, K, Kpad, ldRow, ldCol, total);
        return WB{o, Kpad};
    };
    auto cvt = [&](const float* wsrc, int N, int K) -> WB { return cvtW(wsrc, N, K, K, 1); };

    auto gemmL = [&](const float* A, WB B, const float* bias, float* C,
                     int M, int N, int K, int act, int beta,
                     int lda = -1, int Tdim = 0, int tshift = 0) {
        if (lda < 0) lda = K;
        dim3 g(ceilDiv(N, GBN), ceilDiv(M, GBM));
        gemm_wmma_kernel<<<g, 256, 0, stream>>>(A, B.p, bias, C, M, N, K,
                                                lda, B.ldb, Tdim, tshift, act, beta);
    };

    float* meanB = alloc(DIM);
    float* rstdB = alloc(DIM);
    auto bnL = [&](float* x, int M, int C, const float* g, const float* b, int act) {
        bn_stats_kernel<<<C, 256, 0, stream>>>(x, M, C, meanB, rstdB);
        const int total = M * C;
        bn_apply_kernel<<<ceilDiv(total, 256), 256, 0, stream>>>(
            x, meanB, rstdB, g, b, total, C, act);
    };

    // Stage weights: w0, 4x {3 taps of wd (repacked), w1}
    struct StageW { WB w0; WB tap[4][3]; WB l1[4]; };
    auto makeStage = [&](int base) -> StageW {
        StageW s;
        s.w0 = cvt(P[base], DIM, DIM);
        for (int i = 0; i < 4; ++i) {
            const int lb = base + 2 + 4 * i;
            for (int tap = 0; tap < 3; ++tap)
                s.tap[i][tap] = cvtW(P[lb] + tap, DIM, DIM, DIM * 3, 3);
            s.l1[i] = cvt(P[lb + 2], DIM, DIM);
        }
        return s;
    };
    auto stageRun = [&](const StageW& s, int base, const float* X, float* OUT,
                        float* H, float* T, int M) {
        gemmL(X, s.w0, P[base + 1], OUT, M, DIM, DIM, 0, 0);
        for (int i = 0; i < 4; ++i) {
            const int d = 1 << i, lb = base + 2 + 4 * i;
            for (int tap = 0; tap < 3; ++tap)
                gemmL(OUT, s.tap[i][tap], nullptr, H, M, DIM, DIM, 0, tap > 0,
                      DIM, NFRM, (tap - 1) * d);
            bias_relu_kernel<<<ceilDiv(M * DIM, 256), 256, 0, stream>>>(
                H, P[lb + 1], M * DIM, DIM);
            gemmL(H, s.l1[i], P[lb + 3], T, M, DIM, DIM, 0, 0);
            add_inplace_kernel<<<ceilDiv(M * DIM, 256), 256, 0, stream>>>(
                OUT, T, M * DIM);
        }
    };

    // Convert all linear weights
    WB w_coord1 = cvt(P[1], 128, 4),    w_coord2 = cvt(P[4], DIM, 128);
    WB w_m1a = cvt(P[7], DIM, 512),     w_m1b = cvt(P[9], DIM, DIM);
    WB w_m2a = cvt(P[13], DIM, DIM),    w_m2b = cvt(P[15], DIM, DIM);
    WB w_m3a = cvt(P[19], DIM, 512),    w_m3b = cvt(P[21], DIM, DIM);
    WB w_m4a = cvt(P[25], DIM, 512),    w_m4b = cvt(P[27], DIM, DIM);
    WB w_cvd = cvt(P[34], DIM, DIM),    w_cpd = cvt(P[36], DIM, DIM);
    StageW sV = makeStage(38), sP = makeStage(56), sN = makeStage(74);
    WB w_vf1 = cvt(P[92], DIM, NFRM * DIM),  w_vf2 = cvt(P[95], DIM, DIM);
    WB w_pf1 = cvt(P[98], DIM, NFRM * DIM),  w_pf2 = cvt(P[101], DIM, DIM);
    WB w_nf1 = cvt(P[104], DIM, NFRM * DIM), w_nf2 = cvt(P[105], DIM, DIM);
    WB w_m5a = cvt(P[106], DIM, 512),   w_m5b = cvt(P[109], DIM, DIM);
    WB w_m6a = cvt(P[112], DIM, DIM),   w_m6b = cvt(P[115], DIM, DIM);
    WB w_71  = cvt(P[118], DIM, 512),   w_72  = cvt(P[119], DIM, DIM);
    WB w_c1  = cvt(P[120], DIM, DIM),   w_c2  = cvt(P[122], 512, DIM),
       w_c3  = cvt(P[124], 174, 512);
    WB w_v1  = cvt(P[126], DIM, DIM),   w_v2  = cvt(P[128], 512, DIM),
       w_v3  = cvt(P[130], 61, 512);
    WB w_p1  = cvt(P[132], DIM, DIM),   w_p2  = cvt(P[134], 512, DIM),
       w_p3  = cvt(P[136], 59, 512);

    // ---- activation buffers ---------------------------------------------
    float* x0    = alloc((size_t)MX * 4);
    float* hc    = alloc((size_t)MX * 128);
    float* x     = alloc((size_t)MX * DIM);
    float* eb    = alloc((size_t)ME * 512);
    float* tA    = alloc((size_t)ME * DIM);
    float* tB    = alloc((size_t)ME * DIM);
    float* e4    = alloc((size_t)ME * DIM);
    float* n1    = alloc((size_t)MX * DIM);
    float* nA    = alloc((size_t)MX * DIM);
    float* catn  = alloc((size_t)MX * 512);
    float* n2    = alloc((size_t)MX * DIM);
    float* brMix = alloc((size_t)ME * DIM);
    float* brX   = alloc((size_t)ME * DIM);
    float* brOut = alloc((size_t)ME * DIM);
    float* f1    = alloc((size_t)BATCH * NEDGE * DIM);
    float* f2    = alloc((size_t)BATCH * NEDGE * DIM);
    float* vf    = alloc((size_t)BATCH * DIM);
    float* pf    = alloc((size_t)BATCH * DIM);
    float* nf    = alloc((size_t)BATCH * DIM);
    float* cat2  = alloc((size_t)BATCH * 512);
    float* vp    = alloc((size_t)BATCH * DIM);
    float* gA    = alloc((size_t)BATCH * DIM);
    float* gB    = alloc((size_t)BATCH * DIM);
    float* h512  = alloc((size_t)BATCH * 512);
    float* h256  = alloc((size_t)BATCH * DIM);

    auto nri = [&](int base, WB wa, WB wb, const float* xin, float* outp,
                   float* tmp, int M, int K) {
        gemmL(xin, wa, P[base + 1], tmp, M, DIM, K, 2, 0);
        gemmL(tmp, wb, P[base + 3], outp, M, DIM, DIM, 2, 0);
        bnL(outp, M, DIM, P[base + 4], P[base + 5], 0);
    };
    auto lbr = [&](int base, WB wa, WB wb, const float* xin, float* outp,
                   float* tmp, int M, int K) {
        gemmL(xin, wa, nullptr, tmp, M, DIM, K, 0, 0);
        bnL(tmp, M, DIM, P[base + 1], P[base + 2], 1);
        gemmL(tmp, wb, nullptr, outp, M, DIM, DIM, 0, 0);
        bnL(outp, M, DIM, P[base + 4], P[base + 5], 1);
    };
    auto clsf = [&](const float* in, int base, WB wa, WB wb, WB wc,
                    float* outp, int ncls) {
        gemmL(in, wa, P[base + 1], h256, BATCH, DIM, DIM, 1, 0);
        gemmL(h256, wb, P[base + 3], h512, BATCH, 512, DIM, 1, 0);
        gemmL(h512, wc, P[base + 5], outp, BATCH, ncls, 512, 0, 0);
    };

    // ---- coord encoder ---------------------------------------------------
    transpose_box_kernel<<<ceilDiv(MX * 4, 256), 256, 0, stream>>>(box, x0);
    gemmL(x0, w_coord1, nullptr, hc, MX, 128, 4, 0, 0);
    bnL(hc, MX, 128, P[2], P[3], 1);
    gemmL(hc, w_coord2, nullptr, x, MX, DIM, 128, 0, 0);
    bnL(x, MX, DIM, P[5], P[6], 1);

    // ---- node2edge -> mlp1 -> edge2node -> mlp2 -------------------------
    node2edge_kernel<<<ceilDiv(ME * 512, 256), 256, 0, stream>>>(x, eb, ME * 512);
    nri(7, w_m1a, w_m1b, eb, tA, tB, ME, 512);
    edge2node_kernel<<<ceilDiv(MX * DIM, 256), 256, 0, stream>>>(P[31], tA, nA, MX * DIM);
    nri(13, w_m2a, w_m2b, nA, n1, tB, MX, DIM);

    // ---- skip -> mlp3 -> node2edge -> mlp4 ------------------------------
    concat2_kernel<<<ceilDiv(MX * 512, 256), 256, 0, stream>>>(x, n1, catn, MX, DIM, DIM);
    nri(19, w_m3a, w_m3b, catn, n2, tB, MX, 512);
    node2edge_kernel<<<ceilDiv(ME * 512, 256), 256, 0, stream>>>(n2, eb, ME * 512);
    nri(25, w_m4a, w_m4b, eb, e4, tB, ME, 512);

    // ---- verb edge branch -----------------------------------------------
    bkmix_kernel<<<ceilDiv(ME * DIM, 256), 256, 0, stream>>>(P[32], e4, brMix, ME * DIM);
    gemmL(brMix, w_cvd, P[35], brX, ME, DIM, DIM, 0, 0);
    stageRun(sV, 38, brX, brOut, tA, tB, ME);
    lbr(92, w_vf1, w_vf2, brOut, f2, f1, BATCH * NEDGE, NFRM * DIM);
    mean_group_kernel<<<ceilDiv(BATCH * DIM, 256), 256, 0, stream>>>(f2, vf, BATCH, NEDGE, DIM);

    // ---- preposition edge branch ----------------------------------------
    bkmix_kernel<<<ceilDiv(ME * DIM, 256), 256, 0, stream>>>(P[33], e4, brMix, ME * DIM);
    gemmL(brMix, w_cpd, P[37], brX, ME, DIM, DIM, 0, 0);
    stageRun(sP, 56, brX, brOut, tA, tB, ME);
    lbr(98, w_pf1, w_pf2, brOut, f2, f1, BATCH * NEDGE, NFRM * DIM);
    mean_group_kernel<<<ceilDiv(BATCH * DIM, 256), 256, 0, stream>>>(f2, pf, BATCH, NEDGE, DIM);

    // ---- node temporal branch -------------------------------------------
    stageRun(sN, 74, n2, brOut, tA, tB, MX);
    gemmL(brOut, w_nf1, nullptr, f1, BATCH * NBOX, DIM, NFRM * DIM, 1, 0);
    gemmL(f1, w_nf2, nullptr, f2, BATCH * NBOX, DIM, DIM, 1, 0);
    mean_group_kernel<<<ceilDiv(BATCH * DIM, 256), 256, 0, stream>>>(f2, nf, BATCH, NBOX, DIM);

    // ---- head ------------------------------------------------------------
    concat2_kernel<<<ceilDiv(BATCH * 512, 256), 256, 0, stream>>>(vf, pf, cat2, BATCH, DIM, DIM);
    lbr(106, w_m5a, w_m5b, cat2, vp, h256, BATCH, 512);
    concat2_kernel<<<ceilDiv(BATCH * 512, 256), 256, 0, stream>>>(vp, nf, cat2, BATCH, DIM, DIM);
    gemmL(cat2, w_71, nullptr, gA, BATCH, DIM, 512, 0, 0);
    gemmL(gA, w_72, nullptr, gB, BATCH, DIM, DIM, 0, 0);
    lbr(112, w_m6a, w_m6b, gB, gA, h256, BATCH, DIM);   // gf = gA

    float* outp = (float*)d_out;
    clsf(gA, 120, w_c1, w_c2, w_c3, outp, 174);
    clsf(vf, 126, w_v1, w_v2, w_v3, outp + BATCH * 174, 61);
    clsf(pf, 132, w_p1, w_p2, w_p3, outp + BATCH * (174 + 61), 59);
}
